// QuantLinear_14620068676265
// MI455X (gfx1250) — matrix operand order repaired and checked by reading, not compiled, not run on hardware
//
#include <hip/hip_runtime.h>

typedef __attribute__((ext_vector_type(16))) _Float16 v16h;
typedef __attribute__((ext_vector_type(8)))  _Float16 v8h;
typedef __attribute__((ext_vector_type(2)))  _Float16 h2;
typedef __attribute__((ext_vector_type(8)))  float    v8f;
typedef __attribute__((ext_vector_type(4)))  int      v4i;

#define GROUP_SHIFT 7            // group size 128
#define KSPLIT 8
#define WAVES_PER_BLOCK 8
#define COLS_PER_WAVE 32         // 2 N-tiles of 16
#define COLS_PER_BLOCK (WAVES_PER_BLOCK * COLS_PER_WAVE)   // 256

// Dequant one int32 (one payload byte = 2 nibbles) into two f16 weights.
// 0x6400|nib == fp16(1024+nib); (x - 1031)*s done as one packed FMA.
__device__ __forceinline__ h2 dq2(int qv, h2 s2, h2 c2) {
  unsigned u = (unsigned)qv;
  unsigned packed = 0x64006400u | (u & 0xFu) | ((u & 0xF0u) << 12);
  h2 x = __builtin_bit_cast(h2, packed);
  return __builtin_elementwise_fma(x, s2, c2);
}

// Build a 16-half B fragment (16 consecutive K of one column) from 8 int32s.
__device__ __forceinline__ v16h dq16(v4i q0, v4i q1, _Float16 s) {
  h2 s2 = {s, s};
  _Float16 c = s * (_Float16)(-1031.0f);
  h2 c2 = {c, c};
  v16h b;
  h2 w;
  w = dq2(q0.x, s2, c2); b[0]  = w[0]; b[1]  = w[1];
  w = dq2(q0.y, s2, c2); b[2]  = w[0]; b[3]  = w[1];
  w = dq2(q0.z, s2, c2); b[4]  = w[0]; b[5]  = w[1];
  w = dq2(q0.w, s2, c2); b[6]  = w[0]; b[7]  = w[1];
  w = dq2(q1.x, s2, c2); b[8]  = w[0]; b[9]  = w[1];
  w = dq2(q1.y, s2, c2); b[10] = w[0]; b[11] = w[1];
  w = dq2(q1.z, s2, c2); b[12] = w[0]; b[13] = w[1];
  w = dq2(q1.w, s2, c2); b[14] = w[0]; b[15] = w[1];
  return b;
}

__device__ __forceinline__ v16h cat8(v8h lo, v8h hi) {
  return __builtin_shufflevector(lo, hi, 0,1,2,3,4,5,6,7,8,9,10,11,12,13,14,15);
}

// Partial GEMM over one K-chunk. M fixed at 32 (two 16-row WMMA tiles).
__global__ __launch_bounds__(256) void qgemm_partial(
    const _Float16* __restrict__ A,      // (32, K) f16 row-major
    const int*      __restrict__ qw,     // (N, K/2) int32, one byte payload
    const _Float16* __restrict__ scales, // (N, K/128) f16
    float*          __restrict__ partial,// (KSPLIT, 32, N) f32
    int N, int K, int kChunk)
{
  const int lane  = threadIdx.x & 31;
  const int wave  = threadIdx.x >> 5;
  const int laneN = lane & 15;
  const int sel   = lane >> 4;                 // which K-half this lane owns
  const int nb    = blockIdx.x * COLS_PER_BLOCK + wave * COLS_PER_WAVE;
  const int kc    = blockIdx.y;
  const int kBeg  = kc * kChunk;
  const int kEnd  = kBeg + kChunk;
  const int Kg    = K >> GROUP_SHIFT;
  const int Kq    = K >> 1;

  // A fragment base pointers (this code's contiguous k order: lane m<16 owns K {0..15},
  // lanes 16..31 own K {16..31} of row m-16)
  const _Float16* A0 = A + laneN * K + sel * 16;  // M-tile 0 (rows 0..15)
  const _Float16* A1 = A0 + 16 * K;               // M-tile 1 (rows 16..31)

  // B: lane owns column c, 16 consecutive K starting at k0 + sel*16
  const int c0 = nb + laneN;
  const int c1 = c0 + 16;
  const int* q0 = qw + c0 * Kq + sel * 8;
  const int* q1 = qw + c1 * Kq + sel * 8;
  const _Float16* s0 = scales + c0 * Kg;
  const _Float16* s1 = scales + c1 * Kg;

  v8f acc00 = {}, acc01 = {}, acc10 = {}, acc11 = {};

  for (int k0 = kBeg; k0 < kEnd; k0 += 32) {
    // ---- A fragments (hit L2: A is 512KB, reused by all N-blocks) ----
    v16h a0 = cat8(*(const v8h*)(A0 + k0), *(const v8h*)(A0 + k0 + 8));
    v16h a1 = cat8(*(const v8h*)(A1 + k0), *(const v8h*)(A1 + k0 + 8));

    // ---- weight stream: non-temporal b128 loads (read-once) ----
    const v4i* p0 = (const v4i*)(q0 + (k0 >> 1));
    const v4i* p1 = (const v4i*)(q1 + (k0 >> 1));
    v4i w00 = __builtin_nontemporal_load(p0);
    v4i w01 = __builtin_nontemporal_load(p0 + 1);
    v4i w10 = __builtin_nontemporal_load(p1);
    v4i w11 = __builtin_nontemporal_load(p1 + 1);

    const int g = k0 >> GROUP_SHIFT;   // whole 32-K step sits in one group
    v16h b0 = dq16(w00, w01, s0[g]);
    v16h b1 = dq16(w10, w11, s1[g]);

    acc00 = __builtin_amdgcn_wmma_f32_16x16x32_f16(false, a0, false, b0,
                                                   (short)0, acc00, false, false);
    acc10 = __builtin_amdgcn_wmma_f32_16x16x32_f16(false, a1, false, b0,
                                                   (short)0, acc10, false, false);
    acc01 = __builtin_amdgcn_wmma_f32_16x16x32_f16(false, a0, false, b1,
                                                   (short)0, acc01, false, false);
    acc11 = __builtin_amdgcn_wmma_f32_16x16x32_f16(false, a1, false, b1,
                                                   (short)0, acc11, false, false);
  }

  // D layout: lane L, VGPR j -> row (L>=16 ? 8:0)+j, col L&15
  float* pbase = partial + (size_t)kc * 32 * N;
  const int mrow = sel * 8;
#pragma unroll
  for (int j = 0; j < 8; ++j) {
    pbase[(size_t)(mrow + j)      * N + c0] = acc00[j];
    pbase[(size_t)(mrow + j + 16) * N + c0] = acc10[j];
    pbase[(size_t)(mrow + j)      * N + c1] = acc01[j];
    pbase[(size_t)(mrow + j + 16) * N + c1] = acc11[j];
  }
}

__global__ __launch_bounds__(256) void reduce_bias(
    const float* __restrict__ partial, const _Float16* __restrict__ bias,
    float* __restrict__ out, int MN, int N)
{
  int i = blockIdx.x * 256 + threadIdx.x;
  if (i >= MN) return;
  float s = 0.f;
#pragma unroll
  for (int kc = 0; kc < KSPLIT; ++kc)
    s += partial[(size_t)kc * MN + i];
  out[i] = s + (float)bias[i % N];
}

extern "C" void kernel_launch(void* const* d_in, const int* in_sizes, int n_in,
                              void* d_out, int out_size, void* d_ws, size_t ws_size,
                              hipStream_t stream) {
  const _Float16* A      = (const _Float16*)d_in[0];
  const int*      qw     = (const int*)d_in[1];
  const _Float16* scales = (const _Float16*)d_in[2];
  const _Float16* bias   = (const _Float16*)d_in[3];
  float*          out    = (float*)d_out;
  float*          part   = (float*)d_ws;       // KSPLIT * M * N floats (8 MB)

  const int N = in_sizes[3];                   // 8192
  const int K = (int)(2LL * in_sizes[1] / N);  // 8192
  const int M = in_sizes[0] / K;               // 32 (kernel assumes 32)
  const int kChunk = K / KSPLIT;               // 1024

  dim3 grid(N / COLS_PER_BLOCK, KSPLIT);
  qgemm_partial<<<grid, 256, 0, stream>>>(A, qw, scales, part, N, K, kChunk);

  const int MN = M * N;
  reduce_bias<<<(MN + 255) / 256, 256, 0, stream>>>(part, bias, out, MN, N);
}
